// HOPEBlock_23278722744908
// MI455X (gfx1250) — compile-verified, matrix operand order checked by reading, not run on hardware
//
#include <hip/hip_runtime.h>

#define HIDDEN   1024
#define HEADS    16
#define HEAD_DIM 64
#define INNER    4096
#define BATCH    2
#define SEQ      2048
#define NTOK     (BATCH * SEQ)

typedef __bf16 bf16;
typedef __bf16 v16bf __attribute__((ext_vector_type(16)));
typedef float  v8f   __attribute__((ext_vector_type(8)));
typedef unsigned int v4u __attribute__((ext_vector_type(4)));
typedef int          v8i __attribute__((ext_vector_type(8)));
typedef int          v4i __attribute__((ext_vector_type(4)));

#if __has_builtin(__builtin_amdgcn_tensor_load_to_lds)
#define USE_TDM 1
#else
#define USE_TDM 0
#endif

union FragU { uint4 u[2]; v16bf v; };

__device__ __forceinline__ v16bf load_frag(const bf16* base, int stride, int row, int c0, int c1) {
    FragU f;
    f.u[0] = *reinterpret_cast<const uint4*>(base + (size_t)row * stride + c0);
    f.u[1] = *reinterpret_cast<const uint4*>(base + (size_t)row * stride + c1);
    return f.v;
}

__device__ __forceinline__ v8f wmma_bf16(v16bf a, v16bf b, v8f c) {
    return __builtin_amdgcn_wmma_f32_16x16x32_bf16(false, a, false, b, (short)0, c, false, false);
}

__device__ __forceinline__ unsigned lds_off(const void* p) {
    // generic pointer to LDS: low 32 bits are the wave-relative LDS byte offset
    return (unsigned)(uintptr_t)p;
}

#if USE_TDM
// TDM: DMA one [128 rows x 32 cols] bf16 tile (row stride K elems) from global into LDS,
// writing rows padded to 40 elements (pad_interval = 16 dwords, pad_amount = 4 dwords).
__device__ __forceinline__ void tdm_load_tile_2d(unsigned ldsAddr, const bf16* gptr, int K) {
    unsigned long long ga = (unsigned long long)(uintptr_t)gptr;
    v4u g0;
    g0[0] = 1u;                                           // count=1, user descriptor
    g0[1] = ldsAddr;                                      // lds_addr
    g0[2] = (unsigned)(ga & 0xffffffffull);               // global_addr[31:0]
    g0[3] = (unsigned)((ga >> 32) & 0x1ffffffull) | (2u << 30); // global_addr[56:32] | type=2

    const unsigned td0 = 32, td1 = 128, tile0 = 32, tile1 = 128;
    unsigned long long s0 = (unsigned long long)(unsigned)K;    // tensor_dim0_stride (elems)
    v8i g1;
    g1[0] = (1 << 16)      // data_size = 2 bytes
          | (1 << 20)      // pad_enable
          | (3 << 22)      // pad_interval: 16 dwords (one 32-elem bf16 row)
          | (3 << 25);     // pad_amount: 4 dwords (8 bf16 elems)
    g1[1] = (int)((td0 & 0xffffu) << 16);                     // [79:48] tensor_dim0 (lo)
    g1[2] = (int)((td0 >> 16) | ((td1 & 0xffffu) << 16));     // tensor_dim0 hi | tensor_dim1 lo
    g1[3] = (int)((td1 >> 16) | (tile0 << 16));               // tensor_dim1 hi | tile_dim0
    g1[4] = (int)(tile1);                                     // tile_dim1 | tile_dim2=0
    g1[5] = (int)(s0 & 0xffffffffull);                        // stride0[31:0]
    g1[6] = (int)((s0 >> 32) & 0xffffull);                    // stride0[47:32] | stride1 lo=0
    g1[7] = 0;
    v4i gz4 = {0, 0, 0, 0};
    v8i gz8 = {0, 0, 0, 0, 0, 0, 0, 0};
    // 6-arg form (clang-23 / therock-10.0 headers): groups 0..3 + extra group + cpol
    __builtin_amdgcn_tensor_load_to_lds(g0, g1, gz4, gz4, gz8, 0);
}
#endif

// per-lane async global -> LDS copy of 16 bytes (ASYNCcnt)
__device__ __forceinline__ void async_copy_b128(unsigned ldsAddr, const void* gaddr) {
    asm volatile("global_load_async_to_lds_b128 %0, %1, off"
                 :: "v"(ldsAddr), "v"((unsigned long long)(uintptr_t)gaddr)
                 : "memory");
}
__device__ __forceinline__ void wait_asynccnt0() {
    asm volatile("s_wait_asynccnt 0x0" ::: "memory");
}

// ---------------------------------------------------------------- f32 -> bf16
__global__ __launch_bounds__(256) void cvt_f32_bf16(const float* __restrict__ s,
                                                    bf16* __restrict__ d, int n) {
    int i = blockIdx.x * blockDim.x + threadIdx.x;
    int stride = gridDim.x * blockDim.x;
    for (; i < n; i += stride) d[i] = (bf16)s[i];
}

// ---------------------------------------------------------------- GEMM  C = A * B^T (+bias)(+silu)(+resid)
// A: [M,K] bf16 row-major, B: [N,K] bf16 row-major. Block tile 128x128, BK=32,
// double-buffered LDS fed by the Tensor Data Mover (waves 0/1 issue descriptors).
template<bool HAS_BIAS, bool HAS_RES, bool ACT_SILU, bool OUT_BF16>
__global__ __launch_bounds__(256) void gemm_bf16_wmma(
    const bf16* __restrict__ A, const bf16* __restrict__ Bm,
    const float* __restrict__ bias, const float* __restrict__ resid,
    float* __restrict__ outF, bf16* __restrict__ outB,
    int M, int N, int K)
{
    __shared__ __align__(16) bf16 sA[2][128][40];
    __shared__ __align__(16) bf16 sB[2][128][40];

    const int tid   = threadIdx.x;
    const int lane  = tid & 31;
    const int wave  = tid >> 5;
    const int waveM = wave >> 1;   // 0..3
    const int waveN = wave & 1;    // 0..1
    const int lrow  = lane & 15;
    const int khalf = lane >> 4;
    const int m0 = blockIdx.y * 128;
    const int n0 = blockIdx.x * 128;

    const bf16* Abase = A  + (size_t)m0 * K;
    const bf16* Bbase = Bm + (size_t)n0 * K;

    v8f zero = {0.f,0.f,0.f,0.f,0.f,0.f,0.f,0.f};
    v8f acc[2][4];
#pragma unroll
    for (int mi = 0; mi < 2; ++mi)
#pragma unroll
        for (int ni = 0; ni < 4; ++ni) acc[mi][ni] = zero;

    const int c0 = khalf * 8;
    const int c1 = 16 + khalf * 8;

#if USE_TDM
    // prologue: prefetch tile k=0 into buffer 0
    if (wave == 0)      tdm_load_tile_2d(lds_off(&sA[0][0][0]), Abase, K);
    else if (wave == 1) tdm_load_tile_2d(lds_off(&sB[0][0][0]), Bbase, K);

    int buf = 0;
    for (int k0 = 0; k0 < K; k0 += 32) {
        if (k0 + 32 < K) {   // prefetch next tile into the other buffer (just vacated)
            if (wave == 0)      tdm_load_tile_2d(lds_off(&sA[buf ^ 1][0][0]), Abase + k0 + 32, K);
            else if (wave == 1) tdm_load_tile_2d(lds_off(&sB[buf ^ 1][0][0]), Bbase + k0 + 32, K);
            if (wave < 2) __builtin_amdgcn_s_wait_tensorcnt(1);  // current tile's DMA done
        } else {
            if (wave < 2) __builtin_amdgcn_s_wait_tensorcnt(0);
        }
        __syncthreads();     // current buffer visible to all waves

        v16bf af[2], bfrag[4];
#pragma unroll
        for (int mi = 0; mi < 2; ++mi)
            af[mi] = load_frag(&sA[buf][0][0], 40, waveM * 32 + mi * 16 + lrow, c0, c1);
#pragma unroll
        for (int ni = 0; ni < 4; ++ni)
            bfrag[ni] = load_frag(&sB[buf][0][0], 40, waveN * 64 + ni * 16 + lrow, c0, c1);

#pragma unroll
        for (int mi = 0; mi < 2; ++mi)
#pragma unroll
            for (int ni = 0; ni < 4; ++ni)
                acc[mi][ni] = wmma_bf16(af[mi], bfrag[ni], acc[mi][ni]);

        __syncthreads();     // everyone done reading buf before next prefetch overwrites it
        buf ^= 1;
    }
#else
    const int ldRow = tid >> 1;
    const int ldCol = (tid & 1) * 16;
    for (int k0 = 0; k0 < K; k0 += 32) {
        __syncthreads();
        {
            const uint4* ga = reinterpret_cast<const uint4*>(A + (size_t)(m0 + ldRow) * K + k0 + ldCol);
            uint4* la = reinterpret_cast<uint4*>(&sA[0][ldRow][ldCol]);
            la[0] = ga[0]; la[1] = ga[1];
            const uint4* gb = reinterpret_cast<const uint4*>(Bm + (size_t)(n0 + ldRow) * K + k0 + ldCol);
            uint4* lb = reinterpret_cast<uint4*>(&sB[0][ldRow][ldCol]);
            lb[0] = gb[0]; lb[1] = gb[1];
        }
        __syncthreads();
        v16bf af[2], bfrag[4];
#pragma unroll
        for (int mi = 0; mi < 2; ++mi)
            af[mi] = load_frag(&sA[0][0][0], 40, waveM * 32 + mi * 16 + lrow, c0, c1);
#pragma unroll
        for (int ni = 0; ni < 4; ++ni)
            bfrag[ni] = load_frag(&sB[0][0][0], 40, waveN * 64 + ni * 16 + lrow, c0, c1);
#pragma unroll
        for (int mi = 0; mi < 2; ++mi)
#pragma unroll
            for (int ni = 0; ni < 4; ++ni)
                acc[mi][ni] = wmma_bf16(af[mi], bfrag[ni], acc[mi][ni]);
    }
#endif

    // Epilogue: C layout — VGPR r holds row (khalf*8 + r), lane&15 holds column.
#pragma unroll
    for (int mi = 0; mi < 2; ++mi) {
#pragma unroll
        for (int ni = 0; ni < 4; ++ni) {
            const int col = n0 + waveN * 64 + ni * 16 + lrow;
#pragma unroll
            for (int r = 0; r < 8; ++r) {
                const int row = m0 + waveM * 32 + mi * 16 + khalf * 8 + r;
                float vv = acc[mi][ni][r];
                if (HAS_BIAS) vv += bias[col];
                if (ACT_SILU) vv = vv * (1.0f / (1.0f + __expf(-vv)));
                if (HAS_RES)  vv += resid[(size_t)row * N + col];
                const size_t oidx = (size_t)row * N + col;
                if (OUT_BF16) outB[oidx] = (bf16)vv;
                else          outF[oidx] = vv;
            }
        }
    }
}

// ---------------------------------------------------------------- QKV split + RoPE -> bf16 [B][HEADS][S][HD]
__global__ __launch_bounds__(256) void qkv_rope_split(
    const float* __restrict__ qkv, bf16* __restrict__ q,
    bf16* __restrict__ k, bf16* __restrict__ v)
{
    int idx = blockIdx.x * blockDim.x + threadIdx.x;
    const int i = idx & 31; idx >>= 5;
    const int h = idx & 15; idx >>= 4;
    const int t = idx % 3;  idx /= 3;
    const int s = idx % SEQ;
    const int b = idx / SEQ;
    if (b >= BATCH) return;

    const size_t src = ((((size_t)(b * SEQ + s) * 3 + t) * HEADS + h) * HEAD_DIM) + 2 * i;
    const float x1 = qkv[src], x2 = qkv[src + 1];
    const size_t dst = (((size_t)(b * HEADS + h) * SEQ + s) * HEAD_DIM) + 2 * i;

    if (t == 2) { v[dst] = (bf16)x1; v[dst + 1] = (bf16)x2; return; }

    const float inv = __expf((float)(2 * i) * (-9.210340371976184f / 64.0f));
    const float ang = (float)s * inv;
    float sn, cs; __sincosf(ang, &sn, &cs);
    bf16* o = (t == 0) ? q : k;
    o[dst]     = (bf16)(x1 * cs - x2 * sn);
    o[dst + 1] = (bf16)(x1 * sn + x2 * cs);
}

// ---------------------------------------------------------------- Flash attention
// grid (S/64, HEADS, B), 128 threads = 4 waves; wave w owns 16 query rows.
// K tile is staged with async global->LDS (ASYNCcnt); V is transposed manually.
__global__ __launch_bounds__(128) void flash_attn(
    const bf16* __restrict__ Q, const bf16* __restrict__ Kmat,
    const bf16* __restrict__ Vmat, bf16* __restrict__ O)
{
    __shared__ __align__(16) bf16 sK[64][72];        // [key][hd]
    __shared__ __align__(16) bf16 sV[64][72];        // transposed: [hd][key]
    __shared__ __align__(16) bf16 sP[4][16][72];     // per-wave P tile [m][key]

    const int tid   = threadIdx.x;
    const int lane  = tid & 31;
    const int wave  = tid >> 5;
    const int lrow  = lane & 15;
    const int khalf = lane >> 4;
    const int b  = blockIdx.z;
    const int h  = blockIdx.y;
    const int q0 = blockIdx.x * 64;

    const bf16* qh = Q    + (size_t)(b * HEADS + h) * SEQ * HEAD_DIM;
    const bf16* kh = Kmat + (size_t)(b * HEADS + h) * SEQ * HEAD_DIM;
    const bf16* vh = Vmat + (size_t)(b * HEADS + h) * SEQ * HEAD_DIM;

    const int c0 = khalf * 8;
    const int c1 = 16 + khalf * 8;

    v16bf qf[2];
#pragma unroll
    for (int kk = 0; kk < 2; ++kk)
        qf[kk] = load_frag(qh, HEAD_DIM, q0 + wave * 16 + lrow, kk * 32 + c0, kk * 32 + c1);

    v8f zero = {0.f,0.f,0.f,0.f,0.f,0.f,0.f,0.f};
    v8f accO[4];
#pragma unroll
    for (int ni = 0; ni < 4; ++ni) accO[ni] = zero;
    float mrow[8], lsum[8];
#pragma unroll
    for (int r = 0; r < 8; ++r) { mrow[r] = -1e30f; lsum[r] = 0.f; }

    const int ldR = tid >> 1;           // 0..63 key/seq row
    const int ldC = (tid & 1) * 32;     // 0 / 32 hd offset

    for (int j0 = 0; j0 < SEQ; j0 += 64) {
        __syncthreads();
        {
            // K tile: per-lane async DMA, 4 x 16B per thread, never touches VGPRs
            const bf16* gk = kh + (size_t)(j0 + ldR) * HEAD_DIM + ldC;
            unsigned lk = lds_off(&sK[ldR][ldC]);
#pragma unroll
            for (int c = 0; c < 4; ++c)
                async_copy_b128(lk + c * 16, gk + c * 8);
            // V tile: load + transpose into LDS
            const bf16* gv = vh + (size_t)(j0 + ldR) * HEAD_DIM + ldC;
#pragma unroll
            for (int jj = 0; jj < 32; ++jj) sV[ldC + jj][ldR] = gv[jj];
            wait_asynccnt0();
        }
        __syncthreads();

        // Scores: S = Q * K^T  (scale 1/sqrt(64))
        v8f st[4];
#pragma unroll
        for (int ni = 0; ni < 4; ++ni) {
            v16bf kf0 = load_frag(&sK[0][0], 72, ni * 16 + lrow, c0, c1);
            v16bf kf1 = load_frag(&sK[0][0], 72, ni * 16 + lrow, 32 + c0, 32 + c1);
            v8f a = zero;
            a = wmma_bf16(qf[0], kf0, a);
            a = wmma_bf16(qf[1], kf1, a);
            st[ni] = a * 0.125f;
        }

        // Online softmax
        float nm[8], corr[8];
#pragma unroll
        for (int r = 0; r < 8; ++r) {
            float m = fmaxf(fmaxf(st[0][r], st[1][r]), fmaxf(st[2][r], st[3][r]));
#pragma unroll
            for (int sh = 1; sh < 16; sh <<= 1) m = fmaxf(m, __shfl_xor(m, sh, 32));
            nm[r]   = fmaxf(mrow[r], m);
            corr[r] = __expf(mrow[r] - nm[r]);
            mrow[r] = nm[r];
        }
        float ls[8] = {0.f,0.f,0.f,0.f,0.f,0.f,0.f,0.f};
#pragma unroll
        for (int ni = 0; ni < 4; ++ni)
#pragma unroll
            for (int r = 0; r < 8; ++r) {
                float p = __expf(st[ni][r] - nm[r]);
                ls[r] += p;
                sP[wave][khalf * 8 + r][ni * 16 + lrow] = (bf16)p;
            }
#pragma unroll
        for (int r = 0; r < 8; ++r) {
            float s = ls[r];
#pragma unroll
            for (int sh = 1; sh < 16; sh <<= 1) s += __shfl_xor(s, sh, 32);
            lsum[r] = lsum[r] * corr[r] + s;
        }
#pragma unroll
        for (int ni = 0; ni < 4; ++ni)
#pragma unroll
            for (int r = 0; r < 8; ++r) accO[ni][r] *= corr[r];
        __syncthreads();

        // O += P * V
        v16bf pf[2];
#pragma unroll
        for (int kk = 0; kk < 2; ++kk)
            pf[kk] = load_frag(&sP[wave][0][0], 72, lrow, kk * 32 + c0, kk * 32 + c1);
#pragma unroll
        for (int ni = 0; ni < 4; ++ni) {
            v16bf vf0 = load_frag(&sV[0][0], 72, ni * 16 + lrow, c0, c1);
            v16bf vf1 = load_frag(&sV[0][0], 72, ni * 16 + lrow, 32 + c0, 32 + c1);
            accO[ni] = wmma_bf16(pf[0], vf0, accO[ni]);
            accO[ni] = wmma_bf16(pf[1], vf1, accO[ni]);
        }
    }

#pragma unroll
    for (int ni = 0; ni < 4; ++ni)
#pragma unroll
        for (int r = 0; r < 8; ++r) {
            const int qr = q0 + wave * 16 + khalf * 8 + r;
            O[(size_t)(b * SEQ + qr) * HIDDEN + h * HEAD_DIM + ni * 16 + lrow] =
                (bf16)(accO[ni][r] / lsum[r]);
        }
}

// ---------------------------------------------------------------- RMS over H, scale by norm_w -> bf16
__global__ __launch_bounds__(256) void rms_norm_scale(
    const float* __restrict__ mixed, const float* __restrict__ nw, bf16* __restrict__ y)
{
    __shared__ float red[8];
    const int t = blockIdx.x;
    const float* row = mixed + (size_t)t * HIDDEN;
    float ss = 0.f;
    for (int j = threadIdx.x; j < HIDDEN; j += 256) { float v = row[j]; ss += v * v; }
#pragma unroll
    for (int sh = 1; sh < 32; sh <<= 1) ss += __shfl_xor(ss, sh, 32);
    if ((threadIdx.x & 31) == 0) red[threadIdx.x >> 5] = ss;
    __syncthreads();
    float tot = 0.f;
#pragma unroll
    for (int w = 0; w < 8; ++w) tot += red[w];
    const float r = rsqrtf(tot * (1.0f / HIDDEN) + 1.1920929e-07f);
    for (int j = threadIdx.x; j < HIDDEN; j += 256)
        y[(size_t)t * HIDDEN + j] = (bf16)(row[j] * r * nw[j]);
}

// ----------------------------------------------------------------
extern "C" void kernel_launch(void* const* d_in, const int* in_sizes, int n_in,
                              void* d_out, int out_size, void* d_ws, size_t ws_size,
                              hipStream_t stream)
{
    const float* x      = (const float*)d_in[0];
    const float* qkv_w  = (const float*)d_in[1];
    const float* out_w  = (const float*)d_in[2];
    const float* fc1_w  = (const float*)d_in[3];
    const float* fc1_b  = (const float*)d_in[4];
    const float* fc2_w  = (const float*)d_in[5];
    const float* fc2_b  = (const float*)d_in[6];
    const float* norm_w = (const float*)d_in[7];
    const float* upd_w  = (const float*)d_in[8];
    const float* upd_b  = (const float*)d_in[9];
    const float* sc_w   = (const float*)d_in[10];
    const float* sc_b   = (const float*)d_in[11];

    unsigned char* ws = (unsigned char*)d_ws;
    size_t off = 0;
    auto take = [&](size_t bytes) -> unsigned char* {
        unsigned char* p = ws + off;
        off += (bytes + 255) & ~(size_t)255;
        return p;
    };

    bf16* qkvw_b = (bf16*)take((size_t)3 * HIDDEN * HIDDEN * 2);
    bf16* outw_b = (bf16*)take((size_t)HIDDEN * HIDDEN * 2);
    bf16* fc1w_b = (bf16*)take((size_t)INNER * HIDDEN * 2);
    bf16* fc2w_b = (bf16*)take((size_t)HIDDEN * INNER * 2);
    bf16* updw_b = (bf16*)take((size_t)HIDDEN * HIDDEN * 2);
    bf16* scw_b  = (bf16*)take((size_t)HIDDEN * HIDDEN * 2);
    bf16* xb     = (bf16*)take((size_t)NTOK * HIDDEN * 2);   // x bf16, reused as attn out
    bf16* h_b    = (bf16*)take((size_t)NTOK * HIDDEN * 2);
    unsigned char* regP = take((size_t)NTOK * 3 * HIDDEN * 4);  // qkv f32 -> act_b + mixed
    float* qkv_f = (float*)regP;
    bf16*  act_b = (bf16*)regP;
    float* mixed = (float*)(regP + (size_t)NTOK * INNER * 2);
    unsigned char* regQ = take((size_t)3 * NTOK * HIDDEN * 2);  // q/k/v -> y, z
    bf16* q_b = (bf16*)regQ;
    bf16* k_b = (bf16*)(regQ + (size_t)NTOK * HIDDEN * 2);
    bf16* v_b = (bf16*)(regQ + (size_t)2 * NTOK * HIDDEN * 2);
    bf16* y_b = q_b;
    bf16* z_b = k_b;

    const dim3 blk(256);
    auto cvt = [&](const float* s, bf16* d, size_t n) {
        cvt_f32_bf16<<<dim3((unsigned)((n + 1023) / 1024)), blk, 0, stream>>>(s, d, (int)n);
    };

    cvt(x,     xb,     (size_t)NTOK * HIDDEN);
    cvt(qkv_w, qkvw_b, (size_t)3 * HIDDEN * HIDDEN);
    cvt(out_w, outw_b, (size_t)HIDDEN * HIDDEN);
    cvt(fc1_w, fc1w_b, (size_t)INNER * HIDDEN);
    cvt(fc2_w, fc2w_b, (size_t)HIDDEN * INNER);
    cvt(upd_w, updw_b, (size_t)HIDDEN * HIDDEN);
    cvt(sc_w,  scw_b,  (size_t)HIDDEN * HIDDEN);

    gemm_bf16_wmma<false, false, false, false>
        <<<dim3(3 * HIDDEN / 128, NTOK / 128), blk, 0, stream>>>(
            xb, qkvw_b, nullptr, nullptr, qkv_f, nullptr, NTOK, 3 * HIDDEN, HIDDEN);

    qkv_rope_split<<<dim3(BATCH * SEQ * 3 * HEADS * 32 / 256), blk, 0, stream>>>(qkv_f, q_b, k_b, v_b);

    flash_attn<<<dim3(SEQ / 64, HEADS, BATCH), dim3(128), 0, stream>>>(q_b, k_b, v_b, xb);

    gemm_bf16_wmma<false, true, false, true>
        <<<dim3(HIDDEN / 128, NTOK / 128), blk, 0, stream>>>(
            xb, outw_b, nullptr, x, nullptr, h_b, NTOK, HIDDEN, HIDDEN);

    gemm_bf16_wmma<true, false, true, true>
        <<<dim3(INNER / 128, NTOK / 128), blk, 0, stream>>>(
            h_b, fc1w_b, fc1_b, nullptr, nullptr, act_b, NTOK, INNER, HIDDEN);

    gemm_bf16_wmma<true, false, false, false>
        <<<dim3(HIDDEN / 128, NTOK / 128), blk, 0, stream>>>(
            act_b, fc2w_b, fc2_b, nullptr, mixed, nullptr, NTOK, HIDDEN, INNER);

    rms_norm_scale<<<dim3(NTOK), blk, 0, stream>>>(mixed, norm_w, y_b);

    gemm_bf16_wmma<true, true, false, true>
        <<<dim3(HIDDEN / 128, NTOK / 128), blk, 0, stream>>>(
            y_b, updw_b, upd_b, mixed, nullptr, z_b, NTOK, HIDDEN, HIDDEN);

    gemm_bf16_wmma<true, true, false, false>
        <<<dim3(HIDDEN / 128, NTOK / 128), blk, 0, stream>>>(
            z_b, scw_b, sc_b, x, (float*)d_out, nullptr, NTOK, HIDDEN, HIDDEN);
}
